// RNNnet_48825188221156
// MI455X (gfx1250) — matrix operand order repaired and checked by reading, not compiled, not run on hardware
//
#include <hip/hip_runtime.h>
#include <hip/hip_bf16.h>
#include <math.h>

// ---------------------------------------------------------------------------
// BiLSTM classifier for MI455X (gfx1250, wave32, WMMA 16x16x32 bf16 / f32 acc)
//
// Dims: VOCAB=30000, D_EMB=256, H=512 (4H=2048 gates), T=256, B=128, NCLS=5.
//
//   1) embed_bf16 : gather emb[x], PAD->0, store x0 as (T,B,256) bf16
//   2) pack_w     : repack Wih/Whh into per-lane WMMA B-fragment order (bf16)
//   3) gemm_in<K> : XG = X @ W^T + bih via v_wmma_f32_16x16x32_bf16
//   4) lstm_rec   : persistent per-(dir,batch-slice) kernel; 256 timesteps,
//                   h carry in LDS (bf16), c carry in registers (f32),
//                   4 gates per wave -> cell update fully in-register.
//                   Weight loads are kept *inside* the timestep loop (they
//                   stream from the 192MB L2, 8MB working set) by giving the
//                   base pointer an opaque per-iteration zero — otherwise LLVM
//                   hoists 256 loop-invariant fragment loads and spills them
//                   to scratch (observed in round 2).
//   5) classify   : tiny (128x1024)@(1024x5) head in plain f32 VALU
//
// Roofline: ~620 GFLOP. Input GEMMs (344 GF) run at WMMA rate, fully
// parallel; XG traffic 256MB ~ 11us @ 23.3 TB/s. The recurrence is a 512-step
// dependency chain: whh stays L2-resident, h/c never leave the WGP;
// per-step cost ~256 WMMA/wave + L2 loads hidden by prefetch/pipelining.
// ---------------------------------------------------------------------------

#define T_SEQ   256
#define B_BATCH 128
#define H_DIM   512
#define NG      2048      // 4*H gate width
#define DEMB    256
#define NCLS    5
#define VOCAB   30000
#define PADTOK  (VOCAB - 1)

typedef __bf16 bf16;
typedef __attribute__((ext_vector_type(16))) __bf16 v16bf;
typedef __attribute__((ext_vector_type(8)))  __bf16 v8bf;
typedef __attribute__((ext_vector_type(8)))  float  v8f;
typedef __attribute__((ext_vector_type(4)))  int    v4i;

#if defined(__has_builtin)
#if __has_builtin(__builtin_amdgcn_global_store_async_from_lds_b128) && \
    __has_builtin(__builtin_amdgcn_s_wait_asynccnt)
#define HAVE_ASYNC_LDS 1
#endif
#endif

__device__ __forceinline__ float sigm(float x) { return 1.0f / (1.0f + expf(-x)); }

// Fresh wave-uniform zero every call site/iteration; the compiler cannot
// prove it is zero, so pointer arithmetic using it is never loop-invariant.
__device__ __forceinline__ int opaque_zero() {
    int z;
    asm volatile("s_mov_b32 %0, 0" : "=s"(z));
    return z;
}

// Build a 16x32 bf16 A-fragment from two 16-byte chunks.
// Lane L (m = L&15, half = L>>4) holds K = half*8 + [0..7] in elems 0..7 and
// K = 16 + half*8 + [0..7] in elems 8..15.
__device__ __forceinline__ v16bf make_a(const bf16* p) {
    v8bf a0 = *(const v8bf*)p;
    v8bf a1 = *(const v8bf*)(p + 16);
    v16bf a;
#pragma unroll
    for (int i = 0; i < 8; ++i) { a[i] = a0[i]; a[i + 8] = a1[i]; }
    return a;
}

// --------------------------- 1) embedding gather ---------------------------
__global__ void embed_bf16(const int* __restrict__ x, const float* __restrict__ emb,
                           bf16* __restrict__ x0) {
    size_t idx = (size_t)blockIdx.x * blockDim.x + threadIdx.x;   // T*B*DEMB
    int d = idx % DEMB;
    int b = (idx / DEMB) % B_BATCH;
    int t = idx / (DEMB * B_BATCH);
    int tok = x[b * T_SEQ + t];
    float v = (tok == PADTOK) ? 0.0f : emb[(size_t)tok * DEMB + d];
    x0[idx] = (bf16)v;
}

// --------------------------- 2) weight packing -----------------------------
// W: (NG x K) row-major f32 -> B-fragment tile (nt,kt): lane L supplies
// elems i=0..15 = W[nt*16 + (L&15)][kt*32 + (L>>4)*8 + i + (i>=8 ? 8 : 0)], packed 32B/lane.
__global__ void pack_w(const float* __restrict__ W, bf16* __restrict__ out, int K) {
    int idx  = blockIdx.x * blockDim.x + threadIdx.x;   // (NG/16)*(K/32)*32
    int lane = idx & 31;
    int ktn  = K >> 5;
    int kt   = (idx >> 5) % ktn;
    int nt   = idx / (ktn << 5);
    if (nt >= NG / 16) return;
    int n     = nt * 16 + (lane & 15);
    int kbase = kt * 32 + (lane >> 4) * 8;
    bf16* o = out + ((size_t)(nt * ktn + kt) * 32 + lane) * 16;
#pragma unroll
    for (int i = 0; i < 16; ++i) o[i] = (bf16)W[(size_t)n * K + kbase + i + (i >= 8 ? 8 : 0)];
}

// ------------------- 3) input-to-gate GEMM (WMMA bf16) ---------------------
// C(M=32768, N=2048) = X(MxK) @ W^T + bias, output bf16 in C-fragment order.
// 8 waves/block, 16x64 per wave; K templated so offsets become immediates.
template <int K>
__global__ void __launch_bounds__(256)
gemm_in(const bf16* __restrict__ X, const bf16* __restrict__ Wp,
        const float* __restrict__ bias, bf16* __restrict__ XG) {
    constexpr int KTN = K / 32;
    const int lane = threadIdx.x & 31;
    const int wid  = threadIdx.x >> 5;
    const int gw   = blockIdx.x * 8 + wid;
    const int mt   = gw >> 5;           // 0..2047
    const int ng   = gw & 31;           // group of 4 n-tiles
    const int m    = lane & 15, half = lane >> 4;

    const char* xa = (const char*)X + ((size_t)(mt * 16 + m) * K + half * 8) * 2;
    const char* wb = (const char*)Wp + (size_t)ng * 4 * KTN * 1024 + lane * 32;

    v8f acc[4] = {};
#pragma unroll
    for (int kt = 0; kt < KTN; ++kt) {
        v16bf a = make_a((const bf16*)(xa + kt * 64));
#pragma unroll
        for (int j = 0; j < 4; ++j) {
            v16bf b = *(const v16bf*)(wb + j * (KTN * 1024) + kt * 1024);
            acc[j] = __builtin_amdgcn_wmma_f32_16x16x32_bf16(
                false, a, false, b, (short)0, acc[j], false, false);
        }
    }
    char* ob = (char*)XG + (size_t)mt * 65536 + ng * 2048 + lane * 16;
#pragma unroll
    for (int j = 0; j < 4; ++j) {
        float bv = bias[(ng * 4 + j) * 16 + m];
        v8bf o;
#pragma unroll
        for (int r = 0; r < 8; ++r) o[r] = (bf16)(acc[j][r] + bv);
        *(v8bf*)(ob + j * 512) = o;
    }
}

// ----------------------- 4) persistent LSTM recurrence ---------------------
// 16 blocks: dir = blk>>3, batch-slice bs = blk&7 (rows b0..b0+15).
// Wave w owns columns [w*64, w*64+64) of all 4 gates (16 WMMA tiles/step).
__global__ void __launch_bounds__(256, 1)
lstm_rec(const bf16* __restrict__ xg_f, const bf16* __restrict__ xg_r,
         const bf16* __restrict__ whh_f, const bf16* __restrict__ whh_r,
         const float* __restrict__ bhh_f, const float* __restrict__ bhh_r,
         bf16* __restrict__ h_out) {
    __shared__ bf16 h_lds[16][512 + 8];   // +16B pad: conflict-free b128 A loads

    const int lane = threadIdx.x & 31;
    const int w    = threadIdx.x >> 5;    // 8 waves
    const int dir  = blockIdx.x >> 3;
    const int bs   = blockIdx.x & 7;
    const int b0   = bs * 16;
    const bf16*  xg  = dir ? xg_r  : xg_f;
    const bf16*  whh = dir ? whh_r : whh_f;
    const float* bhh = dir ? bhh_r : bhh_f;
    const int m = lane & 15, half = lane >> 4;

    // Byte bases; all remaining tile offsets are compile-time immediates.
    // whh frag bytes: nt*16384 + kt*1024 + lane*32, nt = G*32 + w*4 + cs
    const char* whh_b = (const char*)whh + (size_t)(w * 4) * 16384 + lane * 32;
    // xg frag bytes: mt*65536 + nt*512 + lane*16
    const char* xg_b = (const char*)xg + (size_t)(w * 4) * 512 + lane * 16;

    // copy-phase geometry: wave w drains rows 2w, 2w+1 of h_lds (1KB each)
    const int crow = 2 * w + (lane >> 4);
    const int ccol = (lane & 15) * 32;

    for (int i = threadIdx.x; i < 16 * (512 + 8); i += 256)
        ((bf16*)h_lds)[i] = (bf16)0.0f;

    float c_reg[4][8];
    float h_new[4][8];
    float bvals[4][4];
#pragma unroll
    for (int cs = 0; cs < 4; ++cs) {
#pragma unroll
        for (int r = 0; r < 8; ++r) c_reg[cs][r] = 0.0f;
#pragma unroll
        for (int G = 0; G < 4; ++G)
            bvals[cs][G] = bhh[G * 512 + w * 64 + cs * 16 + (lane & 15)];
    }
    __syncthreads();

    for (int s = 0; s < T_SEQ; ++s) {
        const int t  = dir ? (T_SEQ - 1 - s) : s;   // reverse dir scans backwards
        const int mt = t * 8 + bs;
        const char* xg_step = xg_b + (size_t)mt * 65536;
        // Opaque zero: keeps the 256 weight-fragment loads *inside* this loop
        // (streaming from L2) instead of being hoisted + spilled to scratch.
        const char* whh_s = whh_b + opaque_zero();

        // Prefetch next step's gate pre-activations while we compute.
        if (s + 1 < T_SEQ) {
            const int tn = dir ? (T_SEQ - 2 - s) : (s + 1);
            const char* pf = xg_b + (size_t)(tn * 8 + bs) * 65536;
#pragma unroll
            for (int G = 0; G < 4; ++G)
#pragma unroll
                for (int cs = 0; cs < 4; ++cs)
                    __builtin_prefetch(pf + G * 16384 + cs * 512, 0, 3);
        }

#pragma unroll
        for (int cs = 0; cs < 4; ++cs) {
            v8f acc[4] = {};
#pragma unroll
            for (int kt = 0; kt < 16; ++kt) {
                v16bf a = make_a(&h_lds[m][kt * 32 + half * 8]);
#pragma unroll
                for (int G = 0; G < 4; ++G) {
                    v16bf b = *(const v16bf*)(whh_s + G * 524288 + cs * 16384 + kt * 1024);
                    acc[G] = __builtin_amdgcn_wmma_f32_16x16x32_bf16(
                        false, a, false, b, (short)0, acc[G], false, false);
                }
            }
            v8bf xv[4];
#pragma unroll
            for (int G = 0; G < 4; ++G)
                xv[G] = *(const v8bf*)(xg_step + G * 16384 + cs * 512);
#pragma unroll
            for (int r = 0; r < 8; ++r) {
                float gi = acc[0][r] + (float)xv[0][r] + bvals[cs][0];
                float gf = acc[1][r] + (float)xv[1][r] + bvals[cs][1];
                float gg = acc[2][r] + (float)xv[2][r] + bvals[cs][2];
                float go = acc[3][r] + (float)xv[3][r] + bvals[cs][3];
                float c  = sigm(gf) * c_reg[cs][r] + sigm(gi) * tanhf(gg);
                c_reg[cs][r] = c;
                h_new[cs][r] = sigm(go) * tanhf(c);
            }
        }
        __syncthreads();   // all reads of old h done before overwrite
#ifdef HAVE_ASYNC_LDS
        __builtin_amdgcn_s_wait_asynccnt(0);   // prev step's h drain finished
#endif
#pragma unroll
        for (int cs = 0; cs < 4; ++cs) {
            int j = w * 64 + cs * 16 + (lane & 15);
#pragma unroll
            for (int r = 0; r < 8; ++r)
                h_lds[r + 8 * half][j] = (bf16)h_new[cs][r];
        }
        __syncthreads();

        // Drain h_lds -> h_out in coalesced 16B chunks (async if available).
        bf16* grow = h_out + ((size_t)(t * B_BATCH + b0 + crow)) * 1024 + dir * 512 + ccol;
        bf16* lrow = &h_lds[crow][ccol];
#pragma unroll
        for (int k = 0; k < 4; ++k) {
#ifdef HAVE_ASYNC_LDS
            __builtin_amdgcn_global_store_async_from_lds_b128(
                (__attribute__((address_space(1))) v4i*)(void*)(grow + k * 8),
                (__attribute__((address_space(3))) v4i*)(void*)(lrow + k * 8), 0, 0);
#else
            *(v8bf*)(grow + k * 8) = *(const v8bf*)(lrow + k * 8);
#endif
        }
    }
#ifdef HAVE_ASYNC_LDS
    __builtin_amdgcn_s_wait_asynccnt(0);
#endif
}

// ----------------------------- 5) classifier -------------------------------
__global__ void classify(const bf16* __restrict__ h1, const float* __restrict__ wout,
                         const float* __restrict__ bout, float* __restrict__ out) {
    int idx = blockIdx.x * blockDim.x + threadIdx.x;
    if (idx >= B_BATCH * NCLS) return;
    int b = idx / NCLS, c = idx % NCLS;
    const bf16*  hp = h1 + ((size_t)(T_SEQ - 1) * B_BATCH + b) * 1024;
    const float* wp = wout + (size_t)c * 1024;
    float acc = bout[c];
    for (int j = 0; j < 1024; ++j) acc += (float)hp[j] * wp[j];
    out[idx] = acc;
}

// ---------------------------------------------------------------------------
extern "C" void kernel_launch(void* const* d_in, const int* in_sizes, int n_in,
                              void* d_out, int out_size, void* d_ws, size_t ws_size,
                              hipStream_t stream) {
    (void)in_sizes; (void)n_in; (void)out_size; (void)ws_size;

    const int*   x     = (const int*)  d_in[0];
    const float* emb   = (const float*)d_in[1];
    const float* wih0f = (const float*)d_in[2];
    const float* whh0f = (const float*)d_in[3];
    const float* bih0f = (const float*)d_in[4];
    const float* bhh0f = (const float*)d_in[5];
    const float* wih0r = (const float*)d_in[6];
    const float* whh0r = (const float*)d_in[7];
    const float* bih0r = (const float*)d_in[8];
    const float* bhh0r = (const float*)d_in[9];
    const float* wih1f = (const float*)d_in[10];
    const float* whh1f = (const float*)d_in[11];
    const float* bih1f = (const float*)d_in[12];
    const float* bhh1f = (const float*)d_in[13];
    const float* wih1r = (const float*)d_in[14];
    const float* whh1r = (const float*)d_in[15];
    const float* bih1r = (const float*)d_in[16];
    const float* bhh1r = (const float*)d_in[17];
    const float* wout  = (const float*)d_in[18];
    const float* bout  = (const float*)d_in[19];
    float* out = (float*)d_out;

    // Workspace layout (~440 MB), all 256B-aligned.
    char* ws = (char*)d_ws;
    size_t off = 0;
    auto take = [&](size_t bytes) -> char* {
        char* p = ws + off;
        off += (bytes + 255) & ~(size_t)255;
        return p;
    };
    bf16* x0      = (bf16*)take((size_t)T_SEQ * B_BATCH * DEMB * 2);   // 16 MB
    bf16* h0      = (bf16*)take((size_t)T_SEQ * B_BATCH * 1024 * 2);   // 64 MB
    bf16* h1      = (bf16*)take((size_t)T_SEQ * B_BATCH * 1024 * 2);   // 64 MB
    bf16* xgf     = (bf16*)take((size_t)T_SEQ * B_BATCH * NG * 2);     // 128 MB
    bf16* xgr     = (bf16*)take((size_t)T_SEQ * B_BATCH * NG * 2);     // 128 MB
    bf16* wih0f_p = (bf16*)take((size_t)NG * DEMB * 2);
    bf16* wih0r_p = (bf16*)take((size_t)NG * DEMB * 2);
    bf16* whh0f_p = (bf16*)take((size_t)NG * H_DIM * 2);
    bf16* whh0r_p = (bf16*)take((size_t)NG * H_DIM * 2);
    bf16* wih1f_p = (bf16*)take((size_t)NG * 1024 * 2);
    bf16* wih1r_p = (bf16*)take((size_t)NG * 1024 * 2);
    bf16* whh1f_p = (bf16*)take((size_t)NG * H_DIM * 2);
    bf16* whh1r_p = (bf16*)take((size_t)NG * H_DIM * 2);

    // 1) embedding
    embed_bf16<<<(T_SEQ * B_BATCH * DEMB) / 256, 256, 0, stream>>>(x, emb, x0);

    // 2) weight packing: threads = (NG/16)*(K/32)*32
    pack_w<<<(128 * (DEMB / 32) * 32) / 256, 256, 0, stream>>>(wih0f, wih0f_p, DEMB);
    pack_w<<<(128 * (DEMB / 32) * 32) / 256, 256, 0, stream>>>(wih0r, wih0r_p, DEMB);
    pack_w<<<(128 * (H_DIM / 32) * 32) / 256, 256, 0, stream>>>(whh0f, whh0f_p, H_DIM);
    pack_w<<<(128 * (H_DIM / 32) * 32) / 256, 256, 0, stream>>>(whh0r, whh0r_p, H_DIM);
    pack_w<<<(128 * (1024 / 32) * 32) / 256, 256, 0, stream>>>(wih1f, wih1f_p, 1024);
    pack_w<<<(128 * (1024 / 32) * 32) / 256, 256, 0, stream>>>(wih1r, wih1r_p, 1024);
    pack_w<<<(128 * (H_DIM / 32) * 32) / 256, 256, 0, stream>>>(whh1f, whh1f_p, H_DIM);
    pack_w<<<(128 * (H_DIM / 32) * 32) / 256, 256, 0, stream>>>(whh1r, whh1r_p, H_DIM);

    // 3/4) layer 0
    gemm_in<DEMB><<<8192, 256, 0, stream>>>(x0, wih0f_p, bih0f, xgf);
    gemm_in<DEMB><<<8192, 256, 0, stream>>>(x0, wih0r_p, bih0r, xgr);
    lstm_rec<<<16, 256, 0, stream>>>(xgf, xgr, whh0f_p, whh0r_p, bhh0f, bhh0r, h0);

    // 3/4) layer 1
    gemm_in<1024><<<8192, 256, 0, stream>>>(h0, wih1f_p, bih1f, xgf);
    gemm_in<1024><<<8192, 256, 0, stream>>>(h0, wih1r_p, bih1r, xgr);
    lstm_rec<<<16, 256, 0, stream>>>(xgf, xgr, whh1f_p, whh1r_p, bhh1f, bhh1r, h1);

    // 5) head
    classify<<<(B_BATCH * NCLS + 255) / 256, 256, 0, stream>>>(h1, wout, bout, out);
}
